// UltraOptimizedLSDBlock_61237643707075
// MI455X (gfx1250) — matrix operand order repaired and checked by reading, not compiled, not run on hardware
//
#include <hip/hip_runtime.h>
#include <stdint.h>

#define Bq 8
#define Sq 2048
#define Dq 512
#define HHq 64
#define WWq 32

typedef _Float16 v16h __attribute__((ext_vector_type(16)));
typedef _Float16 v8h  __attribute__((ext_vector_type(8)));
typedef _Float16 v4h  __attribute__((ext_vector_type(4)));
typedef float    v8f  __attribute__((ext_vector_type(8)));

union F8  { v8f v; float f[8]; };
union H8  { v8h v; _Float16 f[8]; };
union A16 { v16h v; v8h h[2]; };

static __device__ __forceinline__ v8f wmma16(v16h a, v16h b, v8f c) {
  // D = A(16x32 f16) * B(32x16 f16) + C(16x16 f32)
  return __builtin_amdgcn_wmma_f32_16x16x32_f16(false, a, false, b, (short)0, c,
                                                false, false);
}

// ---------------------------------------------------------------------------
// Kernel 1: x = sqrt(a)*x_noisy + sqrt(1-a)*noise_pred, stored as f16
// ---------------------------------------------------------------------------
__global__ __launch_bounds__(256)
void prep_kernel(const float* __restrict__ xn, const float* __restrict__ npd,
                 const float* __restrict__ alpha, const int* __restrict__ ts,
                 _Float16* __restrict__ xh) {
  const int idx4 = blockIdx.x * blockDim.x + threadIdx.x;   // float4 index
  const int b = idx4 >> 18;                                 // S*D/4 = 2^18
  const float a  = alpha[ts[b]];
  const float sa = sqrtf(a);
  const float sb = sqrtf(1.0f - a);
  const float4 x = ((const float4*)xn)[idx4];
  const float4 n = ((const float4*)npd)[idx4];
  v4h o;
  o.x = (_Float16)(sa * x.x + sb * n.x);
  o.y = (_Float16)(sa * x.y + sb * n.y);
  o.z = (_Float16)(sa * x.z + sb * n.z);
  o.w = (_Float16)(sa * x.w + sb * n.w);
  ((v4h*)xh)[idx4] = o;
}

// ---------------------------------------------------------------------------
// Kernel 2: flash-attention (two-phase: stats, then PV) fused with
// LayerNorm + GELU + bilinear warp + confidence blend.
// One block = 4 waves = 16 query rows; wave w owns output cols [128w,128w+128)
// ---------------------------------------------------------------------------
__global__ __launch_bounds__(128)
void attn_kernel(const _Float16* __restrict__ xh,
                 const float* __restrict__ gamma, const float* __restrict__ beta,
                 const float* __restrict__ prev, const float* __restrict__ flow,
                 const float* __restrict__ conf, float* __restrict__ out) {
  __shared__ __align__(64) _Float16 qlds[16 * Dq];     // 16 KB Q tile
  __shared__ __align__(64) _Float16 vt[Dq][32];        // 32 KB V tile (transposed)
  __shared__ __align__(64) _Float16 pbuf[4][16 * 32];  // 4 KB  P bounce per wave
  __shared__ float red1[4][16];
  __shared__ float red2[4][16];

  const int tid    = threadIdx.x;
  const int lane   = tid & 31;
  const int wv     = tid >> 5;
  const int half16 = lane & 15;
  const bool hi    = lane >= 16;

  const int g0 = blockIdx.x << 4;          // first global query row
  const int b  = g0 >> 11;                 // batch (S = 2048)
  const float scale = 0.044194173824159216f;  // 1/sqrt(512)

  // ---- stage Q tile (16 rows x 512 f16 = 16 KB) via async copy to LDS ----
  {
    const _Float16* gq = xh + (size_t)g0 * Dq;
#pragma unroll
    for (int i = 0; i < 8; ++i) {
      const int c = tid + i * 128;                           // 0..1023 16B chunks
      unsigned loff = (unsigned)(uintptr_t)&qlds[c * 8];
      unsigned long long ga = (unsigned long long)(uintptr_t)(gq + c * 8);
      asm volatile("global_load_async_to_lds_b128 %0, %1, off"
                   :: "v"(loff), "v"(ga) : "memory");
    }
    asm volatile("s_wait_asynccnt 0" ::: "memory");
  }
  __syncthreads();

  // ======================= Phase A: softmax stats =========================
  float m_run[8], l_run[8];
#pragma unroll
  for (int v = 0; v < 8; ++v) { m_run[v] = -3.0e38f; l_run[v] = 0.0f; }

#pragma unroll 1
  for (int kt = 0; kt < Sq / 32; ++kt) {
    const int kr0 = (b << 11) + kt * 32;
    F8 ct0, ct1;
#pragma unroll
    for (int k = 0; k < 8; ++k) { ct0.f[k] = 0.0f; ct1.f[k] = 0.0f; }

#pragma unroll
    for (int kc = 0; kc < Dq; kc += 32) {
      A16 af;
      const int aoff = kc + (hi ? 16 : 0);
      af.h[0] = *(const v8h*)(qlds + half16 * Dq + aoff);
      af.h[1] = *(const v8h*)(qlds + half16 * Dq + aoff + 8);
      const _Float16* kb0 = xh + (size_t)(kr0 + half16) * Dq + kc + (hi ? 16 : 0);
      const v16h b0 = *(const v16h*)kb0;               // keys n0..n0+15
      const v16h b1 = *(const v16h*)(kb0 + 16 * Dq);   // keys n0+16..n0+31
      ct0.v = wmma16(af.v, b0, ct0.v);
      ct1.v = wmma16(af.v, b1, ct1.v);
    }
#pragma unroll
    for (int v = 0; v < 8; ++v) {
      const float s0 = ct0.f[v] * scale;
      const float s1 = ct1.f[v] * scale;
      float mx = fmaxf(s0, s1);
#pragma unroll
      for (int off = 8; off >= 1; off >>= 1) mx = fmaxf(mx, __shfl_xor(mx, off, 32));
      const float mn = fmaxf(m_run[v], mx);
      float e = __expf(s0 - mn) + __expf(s1 - mn);
#pragma unroll
      for (int off = 8; off >= 1; off >>= 1) e += __shfl_xor(e, off, 32);
      l_run[v] = l_run[v] * __expf(m_run[v] - mn) + e;
      m_run[v] = mn;
    }
  }

  // ======================= Phase B: y = P * V =============================
  F8 acc[8];
#pragma unroll
  for (int j = 0; j < 8; ++j)
#pragma unroll
    for (int k = 0; k < 8; ++k) acc[j].f[k] = 0.0f;

  const int c0 = wv * 128;

#pragma unroll 1
  for (int kt = 0; kt < Sq / 32; ++kt) {
    const int kr0 = (b << 11) + kt * 32;

    __syncthreads();
    // cooperative: stage V tile (32 keys x 512) transposed into vt[col][key]
#pragma unroll
    for (int i = 0; i < 16; ++i) {
      const int cc = i * 128 + tid;         // 0..2047 v8h chunks
      const int k  = cc >> 6;               // key 0..31
      const int c8 = (cc & 63) * 8;         // col chunk
      H8 val; val.v = *(const v8h*)(xh + (size_t)(kr0 + k) * Dq + c8);
#pragma unroll
      for (int j = 0; j < 8; ++j) vt[c8 + j][k] = val.f[j];
    }
    __syncthreads();

    // scores tile (same as phase A)
    F8 ct0, ct1;
#pragma unroll
    for (int k = 0; k < 8; ++k) { ct0.f[k] = 0.0f; ct1.f[k] = 0.0f; }
#pragma unroll
    for (int kc = 0; kc < Dq; kc += 32) {
      A16 af;
      const int aoff = kc + (hi ? 16 : 0);
      af.h[0] = *(const v8h*)(qlds + half16 * Dq + aoff);
      af.h[1] = *(const v8h*)(qlds + half16 * Dq + aoff + 8);
      const _Float16* kb0 = xh + (size_t)(kr0 + half16) * Dq + kc + (hi ? 16 : 0);
      const v16h b0 = *(const v16h*)kb0;
      const v16h b1 = *(const v16h*)(kb0 + 16 * Dq);
      ct0.v = wmma16(af.v, b0, ct0.v);
      ct1.v = wmma16(af.v, b1, ct1.v);
    }

    // P = exp(s - m): write C-layout, read back A-layout (per-wave bounce)
#pragma unroll
    for (int v = 0; v < 8; ++v) {
      const int r = v + (hi ? 8 : 0);
      const float p0 = __expf(ct0.f[v] * scale - m_run[v]);
      const float p1 = __expf(ct1.f[v] * scale - m_run[v]);
      pbuf[wv][r * 32 + half16]      = (_Float16)p0;
      pbuf[wv][r * 32 + 16 + half16] = (_Float16)p1;
    }
    A16 pa;
    const int poff = hi ? 16 : 0;
    pa.h[0] = *(const v8h*)(&pbuf[wv][half16 * 32 + poff]);
    pa.h[1] = *(const v8h*)(&pbuf[wv][half16 * 32 + poff + 8]);

#pragma unroll
    for (int j = 0; j < 8; ++j) {
      const int col = c0 + j * 16 + half16;
      const v16h bv = *(const v16h*)(&vt[col][hi ? 16 : 0]);
      acc[j].v = wmma16(pa.v, bv, acc[j].v);
    }
  }

  // =============== Epilogue: /l, LayerNorm, GELU, warp, blend =============
  F8 yv[8];
#pragma unroll
  for (int j = 0; j < 8; ++j)
#pragma unroll
    for (int v = 0; v < 8; ++v) yv[j].f[v] = acc[j].f[v] / l_run[v];

#pragma unroll
  for (int v = 0; v < 8; ++v) {
    float s = 0.0f, q = 0.0f;
#pragma unroll
    for (int j = 0; j < 8; ++j) { const float t = yv[j].f[v]; s += t; q += t * t; }
#pragma unroll
    for (int off = 8; off >= 1; off >>= 1) {
      s += __shfl_xor(s, off, 32);
      q += __shfl_xor(q, off, 32);
    }
    if (half16 == 0) {
      const int r = v + (hi ? 8 : 0);
      red1[wv][r] = s; red2[wv][r] = q;
    }
  }
  __syncthreads();

  float ga[8], be[8];
#pragma unroll
  for (int j = 0; j < 8; ++j) {
    const int d = c0 + j * 16 + half16;
    ga[j] = gamma[d]; be[j] = beta[d];
  }

#pragma unroll
  for (int v = 0; v < 8; ++v) {
    const int r = v + (hi ? 8 : 0);
    const float tsum = red1[0][r] + red1[1][r] + red1[2][r] + red1[3][r];
    const float tsq  = red2[0][r] + red2[1][r] + red2[2][r] + red2[3][r];
    const float mu   = tsum * (1.0f / Dq);
    const float var  = tsq * (1.0f / Dq) - mu * mu;
    const float rinv = rsqrtf(var + 1e-5f);

    const int srow = (g0 & (Sq - 1)) + r;       // pixel index within batch
    const int h = srow >> 5, w = srow & 31;
    const float fx = flow[((size_t)b * 2 + 0) * (HHq * WWq) + srow];
    const float fy = flow[((size_t)b * 2 + 1) * (HHq * WWq) + srow];
    const float cf = conf[(size_t)b * (HHq * WWq) + srow];
    const float gy = fminf(fmaxf((float)h + fy, 0.0f), (float)(HHq - 1));
    const float gx = fminf(fmaxf((float)w + fx, 0.0f), (float)(WWq - 1));
    const int iy0 = (int)floorf(gy), ix0 = (int)floorf(gx);
    const int iy1 = min(iy0 + 1, HHq - 1), ix1 = min(ix0 + 1, WWq - 1);
    const float wy = gy - (float)iy0, wx = gx - (float)ix0;
    const float w00 = (1.0f - wy) * (1.0f - wx), w01 = (1.0f - wy) * wx;
    const float w10 = wy * (1.0f - wx),          w11 = wy * wx;

#pragma unroll
    for (int j = 0; j < 8; ++j) {
      const int d = c0 + j * 16 + half16;
      const float z  = ga[j] * (yv[j].f[v] - mu) * rinv + be[j];
      const float z3 = z * z * z;
      const float gel = 0.5f * z * (1.0f + tanhf(0.7978845608f * (z + 0.044715f * z3)));
      const size_t pb = ((size_t)b * Dq + d) * (size_t)(HHq * WWq);
      const float warped = prev[pb + iy0 * WWq + ix0] * w00
                         + prev[pb + iy0 * WWq + ix1] * w01
                         + prev[pb + iy1 * WWq + ix0] * w10
                         + prev[pb + iy1 * WWq + ix1] * w11;
      out[pb + srow] = cf * warped + (1.0f - cf) * gel;
    }
  }
}

// ---------------------------------------------------------------------------
extern "C" void kernel_launch(void* const* d_in, const int* in_sizes, int n_in,
                              void* d_out, int out_size, void* d_ws, size_t ws_size,
                              hipStream_t stream) {
  (void)in_sizes; (void)n_in; (void)out_size; (void)ws_size;
  const float* xn    = (const float*)d_in[0];
  const float* npd   = (const float*)d_in[1];
  const float* alpha = (const float*)d_in[2];
  /* d_in[3] = beta_sched (unused by reference) */
  const int*   ts    = (const int*)d_in[4];
  const float* gamma = (const float*)d_in[5];
  const float* beta  = (const float*)d_in[6];
  const float* prev  = (const float*)d_in[7];
  const float* flow  = (const float*)d_in[8];
  const float* conf  = (const float*)d_in[9];
  float* out = (float*)d_out;
  _Float16* xh = (_Float16*)d_ws;   // 16 MB f16 staging of mixed signal x

  prep_kernel<<<(Bq * Sq * Dq / 4) / 256, 256, 0, stream>>>(xn, npd, alpha, ts, xh);
  attn_kernel<<<Bq * Sq / 16, 128, 0, stream>>>(xh, gamma, beta, prev, flow, conf, out);
}
